// Speller_83923660963857
// MI455X (gfx1250) — matrix operand order repaired and checked by reading, not compiled, not run on hardware
//
#include <hip/hip_runtime.h>
#include <hip/hip_bf16.h>

// ---------------------------------------------------------------------------
// Speller decoder step for MI455X (gfx1250, wave32).
//   - LSTM gate GEMMs + output-head GEMM use v_wmma_f32_16x16x32_f16
//     (f16 inputs, f32 accumulate).
//   - Attention (HBM-bound: 1.07 GB of hk/hv streaming, ~46us at 23.3 TB/s)
//     uses coalesced float4 loads + wave32 shuffle reductions.
// ---------------------------------------------------------------------------

#define BB 2048
#define TT 512
#define OO 128
#define EE 512
#define HH 128
#define VV 34

typedef __attribute__((ext_vector_type(16))) _Float16 v16h;
typedef __attribute__((ext_vector_type(8)))  float    v8f;

// ---------------------------------------------------------------------------
// Pack up to three f32 row-major sources into one contiguous f16 matrix
// X[B, wa+wb+wc].  (wc may be 0.)
// ---------------------------------------------------------------------------
__global__ void pack3_f16(const float* __restrict__ a, int wa,
                          const float* __restrict__ b, int wb,
                          const float* __restrict__ c, int wc,
                          _Float16* __restrict__ X, int rows) {
    int K = wa + wb + wc;
    int i = blockIdx.x * blockDim.x + threadIdx.x;
    if (i >= rows * K) return;
    int r = i / K, k = i % K;
    float v;
    if (k < wa)            v = a[(size_t)r * wa + k];
    else if (k < wa + wb)  v = b[(size_t)r * wb + (k - wa)];
    else                   v = c[(size_t)r * wc + (k - wa - wb)];
    X[i] = (_Float16)v;
}

// Pack concatenated weights [Wih | Whh] (row-major, N rows) into f16.
__global__ void packw_f16(const float* __restrict__ Wih, int kin,
                          const float* __restrict__ Whh, int kh,
                          _Float16* __restrict__ W, int N) {
    int K = kin + kh;
    int i = blockIdx.x * blockDim.x + threadIdx.x;
    if (i >= N * K) return;
    int n = i / K, k = i % K;
    float v = (k < kin) ? Wih[(size_t)n * kin + k]
                        : Whh[(size_t)n * kh + (k - kin)];
    W[i] = (_Float16)v;
}

// ---------------------------------------------------------------------------
// C[M,N] = A[M,K] @ W[N,K]^T   (A,W f16 row-major, C f32 row-major)
// One wave computes a 16x16 tile; blockDim.x = 128 (4 waves -> 64 cols).
// Fragment layouts per CDNA5 ISA (16-bit A 16x32, B 32x16, f32 C/D).
// ---------------------------------------------------------------------------
__global__ void gemm_wmma_f16(const _Float16* __restrict__ A,
                              const _Float16* __restrict__ W,
                              float* __restrict__ C,
                              int M, int N, int K) {
    int lane = threadIdx.x & 31;
    int wave = threadIdx.x >> 5;
    int grp  = lane >> 4;        // half-wave group
    int mn   = lane & 15;
    int col0 = (blockIdx.x * 4 + wave) * 16;
    int row0 = blockIdx.y * 16;
    const _Float16* arow = A + (size_t)(row0 + mn) * K;
    const _Float16* wrow = W + (size_t)(col0 + mn) * K;
    v8f acc = {};
    for (int k0 = 0; k0 < K; k0 += 32) {
        union { v16h v; unsigned int u[8]; } af, bf;
#pragma unroll
        for (int j = 0; j < 4; ++j) {
            // A: VGPR j   -> K = grp*8 + 2j,2j+1 ; VGPR j+4 -> +16
            af.u[j]     = *(const unsigned int*)(arow + k0 + grp * 8 + 2 * j);
            af.u[j + 4] = *(const unsigned int*)(arow + k0 + 16 + grp * 8 + 2 * j);
            // B: VGPR j   -> K = grp*8 + 2j,2j+1 ; VGPR j+4 -> +16
            bf.u[j]     = *(const unsigned int*)(wrow + k0 + grp * 8 + 2 * j);
            bf.u[j + 4] = *(const unsigned int*)(wrow + k0 + 16 + grp * 8 + 2 * j);
        }
        acc = __builtin_amdgcn_wmma_f32_16x16x32_f16(
            false, af.v, false, bf.v, (short)0, acc, false, false);
    }
#pragma unroll
    for (int r = 0; r < 8; ++r)   // C/D: VGPR r -> M = r + 8*grp, N = mn
        C[(size_t)(row0 + r + 8 * grp) * N + col0 + mn] = acc[r];
}

// ---------------------------------------------------------------------------
// LSTM elementwise: gates G[B,512] (i,f,g,o blocks of 128) + biases ->
// c_new = sig(f)*c_prev + sig(i)*tanh(g); h_new = sig(o)*tanh(c_new)
// ---------------------------------------------------------------------------
__device__ __forceinline__ float sigf(float x) { return 1.f / (1.f + expf(-x)); }

__global__ void lstm_act(const float* __restrict__ G,
                         const float* __restrict__ bih,
                         const float* __restrict__ bhh,
                         const float* __restrict__ cprev,
                         float* __restrict__ hout, float* __restrict__ cout) {
    int i = blockIdx.x * blockDim.x + threadIdx.x;
    if (i >= BB * OO) return;
    int b = i >> 7, o = i & 127;
    const float* g = G + (size_t)b * 512;
    float ig = g[o]        + bih[o]        + bhh[o];
    float fg = g[128 + o]  + bih[128 + o]  + bhh[128 + o];
    float gg = g[256 + o]  + bih[256 + o]  + bhh[256 + o];
    float og = g[384 + o]  + bih[384 + o]  + bhh[384 + o];
    float cn = sigf(fg) * cprev[i] + sigf(ig) * tanhf(gg);
    hout[i] = sigf(og) * tanhf(cn);
    cout[i] = cn;
}

// ---------------------------------------------------------------------------
// Attention: one block (256 thr = 8 waves) per batch row.
// scores[t] = hk[b,t,:].sc2[b,:]; p = softmax; p*=mask; p/=sum(p)  (the first
// softmax denominator cancels, so: e=exp(s-max)*mask, p=e/sum(e));
// c[o] = sum_t p[t]*hv[b,t,o].  Fully HBM-bound -> coalesced float4 streaming.
// ---------------------------------------------------------------------------
__global__ void attention_kernel(const float* __restrict__ hk,
                                 const float* __restrict__ hv,
                                 const float* __restrict__ q,     // sc2 [B,128]
                                 const float* __restrict__ mask,  // [B,T]
                                 float* __restrict__ cout) {      // [B,128]
    __shared__ float s[TT];
    __shared__ float red[256];
    int b = blockIdx.x, tid = threadIdx.x;
    int lane = tid & 31, wv = tid >> 5;

    float4 qq = reinterpret_cast<const float4*>(q + (size_t)b * OO)[lane];
    const float* hkb = hk + (size_t)b * TT * OO;
    for (int t = wv; t < TT; t += 8) {
        float4 kk = reinterpret_cast<const float4*>(hkb + (size_t)t * OO)[lane];
        float d = kk.x * qq.x + kk.y * qq.y + kk.z * qq.z + kk.w * qq.w;
#pragma unroll
        for (int off = 16; off > 0; off >>= 1) d += __shfl_xor(d, off, 32);
        if (lane == 0) s[t] = d;
    }
    __syncthreads();

    // max over T
    red[tid] = fmaxf(s[tid], s[tid + 256]);
    __syncthreads();
    for (int st = 128; st > 0; st >>= 1) {
        if (tid < st) red[tid] = fmaxf(red[tid], red[tid + st]);
        __syncthreads();
    }
    float mx = red[0];
    __syncthreads();

    // e = exp(s - max) * mask ; sum ; normalize
    float e0 = expf(s[tid] - mx)       * mask[(size_t)b * TT + tid];
    float e1 = expf(s[tid + 256] - mx) * mask[(size_t)b * TT + tid + 256];
    s[tid] = e0; s[tid + 256] = e1;
    red[tid] = e0 + e1;
    __syncthreads();
    for (int st = 128; st > 0; st >>= 1) {
        if (tid < st) red[tid] += red[tid + st];
        __syncthreads();
    }
    float inv = 1.f / red[0];
    __syncthreads();
    s[tid] *= inv; s[tid + 256] *= inv;
    __syncthreads();

    // context: thread -> (o = tid&127, half = tid>>7), coalesced over o
    int o = tid & 127, half = tid >> 7;
    const float* hvb = hv + (size_t)b * TT * OO;
    float acc = 0.f;
    for (int t = half * 256; t < half * 256 + 256; ++t)
        acc += s[t] * hvb[(size_t)t * OO + o];
    red[tid] = acc;
    __syncthreads();
    if (half == 0) cout[(size_t)b * OO + o] = red[o] + red[o + 128];
}

// ---------------------------------------------------------------------------
// BatchNorm stats (training mode, biased var) over B per feature.
// b1/b2 omitted upstream: per-feature constants cancel exactly under BN.
// scale = gamma*rsqrt(var+eps); shift = beta - mu*scale
// ---------------------------------------------------------------------------
__global__ void bn_stats(const float* __restrict__ X,   // [B,H] pre-BN
                         const float* __restrict__ gamma,
                         const float* __restrict__ beta,
                         float* __restrict__ scale, float* __restrict__ shift) {
    __shared__ float ss[256], sq[256];
    int h = blockIdx.x, tid = threadIdx.x;
    float s = 0.f, q = 0.f;
    for (int b = tid; b < BB; b += 256) {
        float v = X[(size_t)b * HH + h];
        s += v; q += v * v;
    }
    ss[tid] = s; sq[tid] = q;
    __syncthreads();
    for (int st = 128; st > 0; st >>= 1) {
        if (tid < st) { ss[tid] += ss[tid + st]; sq[tid] += sq[tid + st]; }
        __syncthreads();
    }
    if (tid == 0) {
        float mu  = ss[0] * (1.f / BB);
        float var = sq[0] * (1.f / BB) - mu * mu;
        float sc  = gamma[h] * rsqrtf(var + 1e-5f);
        scale[h] = sc;
        shift[h] = beta[h] - mu * sc;
    }
}

// ---------------------------------------------------------------------------
// Head: relu(BN(x)) @ w3^T + b3 -> softmax over V=34. One block per batch row.
// ---------------------------------------------------------------------------
__global__ void head_kernel(const float* __restrict__ X,     // [B,H] pre-BN
                            const float* __restrict__ scale,
                            const float* __restrict__ shift,
                            const float* __restrict__ w3,    // [V,H]
                            const float* __restrict__ b3,
                            float* __restrict__ out) {       // [B,V]
    __shared__ float xs[HH];
    __shared__ float lg[VV];
    __shared__ float invsum;
    int b = blockIdx.x, tid = threadIdx.x;   // blockDim = 64
    for (int k = tid; k < HH; k += 64) {
        float v = X[(size_t)b * HH + k] * scale[k] + shift[k];
        xs[k] = fmaxf(v, 0.f);
    }
    __syncthreads();
    if (tid < VV) {
        float a = b3[tid];
        const float* wr = w3 + (size_t)tid * HH;
        for (int k = 0; k < HH; ++k) a += xs[k] * wr[k];
        lg[tid] = a;
    }
    __syncthreads();
    if (tid == 0) {
        float mx = lg[0];
        for (int v = 1; v < VV; ++v) mx = fmaxf(mx, lg[v]);
        float sm = 0.f;
        for (int v = 0; v < VV; ++v) { float e = expf(lg[v] - mx); lg[v] = e; sm += e; }
        invsum = 1.f / sm;
    }
    __syncthreads();
    if (tid < VV) out[(size_t)b * VV + tid] = lg[tid] * invsum;
}

// ---------------------------------------------------------------------------
extern "C" void kernel_launch(void* const* d_in, const int* in_sizes, int n_in,
                              void* d_out, int out_size, void* d_ws, size_t ws_size,
                              hipStream_t stream) {
    const float* hk    = (const float*)d_in[0];
    const float* hv    = (const float*)d_in[1];
    const float* y_1   = (const float*)d_in[2];
    const float* c_1   = (const float*)d_in[3];
    const float* sh_1  = (const float*)d_in[4];   // [3,B,O]
    const float* sc_1  = (const float*)d_in[5];   // [3,B,O]
    const float* mask  = (const float*)d_in[6];
    const float* Wih[3] = { (const float*)d_in[7],  (const float*)d_in[11], (const float*)d_in[15] };
    const float* Whh[3] = { (const float*)d_in[8],  (const float*)d_in[12], (const float*)d_in[16] };
    const float* bih[3] = { (const float*)d_in[9],  (const float*)d_in[13], (const float*)d_in[17] };
    const float* bhh[3] = { (const float*)d_in[10], (const float*)d_in[14], (const float*)d_in[18] };
    const float* w1    = (const float*)d_in[19];
    const float* w2    = (const float*)d_in[21];
    const float* w3    = (const float*)d_in[23];
    const float* b3    = (const float*)d_in[24];
    const float* gamma = (const float*)d_in[25];
    const float* beta  = (const float*)d_in[26];

    // output layout: [out B*V][c B*O][sh 3*B*O][sc 3*B*O]
    float* out_p = (float*)d_out;
    float* c_p   = out_p + (size_t)BB * VV;
    float* sh_p  = c_p + (size_t)BB * OO;
    float* sc_p  = sh_p + 3 * (size_t)BB * OO;

    // workspace layout
    char* w = (char*)d_ws;
    _Float16* Xh = (_Float16*)w;                                    // B x 768 f16
    _Float16* Wh = (_Float16*)(w + (size_t)BB * 768 * 2);           // 512 x 768 f16
    float* G     = (float*)(w + (size_t)BB * 768 * 2 + 512 * 768 * 2); // B x 512 f32
    float* OUT1  = (float*)((char*)G + (size_t)BB * 512 * 4);       // B x 128 f32
    float* bnscale = OUT1 + (size_t)BB * HH;
    float* bnshift = bnscale + HH;

    dim3 b256(256), b128(128), b64(64);

    // ---- LSTM cell 0: X = [y_1 | c_1 | sh_1[0]]  (K = 768) ----
    {
        int n = BB * 768;
        pack3_f16<<<(n + 255) / 256, b256, 0, stream>>>(y_1, EE, c_1, OO, sh_1, OO, Xh, BB);
        int nw = 512 * 768;
        packw_f16<<<(nw + 255) / 256, b256, 0, stream>>>(Wih[0], EE + OO, Whh[0], OO, Wh, 512);
        gemm_wmma_f16<<<dim3(512 / 64, BB / 16), b128, 0, stream>>>(Xh, Wh, G, BB, 512, 768);
        lstm_act<<<(BB * OO + 255) / 256, b256, 0, stream>>>(
            G, bih[0], bhh[0], sc_1, sh_p, sc_p);
    }
    // ---- LSTM cells 1,2: X = [h_below | sh_1[l]]  (K = 256) ----
    for (int l = 1; l < 3; ++l) {
        const float* xin = sh_p + (size_t)(l - 1) * BB * OO;   // new hidden from below
        int n = BB * 256;
        pack3_f16<<<(n + 255) / 256, b256, 0, stream>>>(
            xin, OO, sh_1 + (size_t)l * BB * OO, OO, xin, 0, Xh, BB);
        int nw = 512 * 256;
        packw_f16<<<(nw + 255) / 256, b256, 0, stream>>>(Wih[l], OO, Whh[l], OO, Wh, 512);
        gemm_wmma_f16<<<dim3(512 / 64, BB / 16), b128, 0, stream>>>(Xh, Wh, G, BB, 512, 256);
        lstm_act<<<(BB * OO + 255) / 256, b256, 0, stream>>>(
            G, bih[l], bhh[l], sc_1 + (size_t)l * BB * OO,
            sh_p + (size_t)l * BB * OO, sc_p + (size_t)l * BB * OO);
    }

    // ---- Attention (query = cell state sc2) ----
    attention_kernel<<<BB, b256, 0, stream>>>(hk, hv, sc_p + 2 * (size_t)BB * OO, mask, c_p);

    // ---- Output head GEMM: [sh2 | c] @ [w1 | w2]^T  (b1,b2 cancel under BN) ----
    {
        const float* sh2 = sh_p + 2 * (size_t)BB * OO;
        int n = BB * 256;
        pack3_f16<<<(n + 255) / 256, b256, 0, stream>>>(sh2, OO, c_p, OO, sh2, 0, Xh, BB);
        int nw = HH * 256;
        packw_f16<<<(nw + 255) / 256, b256, 0, stream>>>(w1, OO, w2, OO, Wh, HH);
        gemm_wmma_f16<<<dim3(HH / 64, BB / 16), b128, 0, stream>>>(Xh, Wh, OUT1, BB, HH, 256);
    }

    // ---- BatchNorm stats + head softmax ----
    bn_stats<<<HH, b256, 0, stream>>>(OUT1, gamma, beta, bnscale, bnshift);
    head_kernel<<<BB, b64, 0, stream>>>(OUT1, bnscale, bnshift, w3, b3, out_p);
}
